// GenderGuesser_53206054862916
// MI455X (gfx1250) — compile-verified, matrix operand order checked by reading, not run on hardware
//
#include <hip/hip_runtime.h>
#include <hip/hip_bf16.h>

typedef __attribute__((ext_vector_type(16))) __bf16 v16bf;
typedef __attribute__((ext_vector_type(4)))  __bf16 v4bf;
typedef __attribute__((ext_vector_type(8)))  float  v8f;
typedef __attribute__((ext_vector_type(4)))  int    v4i;

#define SEQ   512
#define HID   2048
#define G3    (3 * HID)       // 6144
#define GRU_BLOCKS 64
#define GRU_TPB    256
#define NWAVES (GRU_BLOCKS * GRU_TPB / 32)   // 512 wave32s
#define JPW    (HID / NWAVES)                // 4 hidden units per wave

#define HAVE_ASYNC_LDS 0
#if defined(__has_builtin)
#if __has_builtin(__builtin_amdgcn_global_load_async_to_lds_b128) && \
    __has_builtin(__builtin_amdgcn_s_wait_asynccnt)
#undef HAVE_ASYNC_LDS
#define HAVE_ASYNC_LDS 1
#endif
#endif

union FragU {
    v16bf v;
    uint4 q[2];
};

// 16B global -> LDS copy, async (gfx1250 GLOBAL_LOAD_ASYNC_TO_LDS_B128) if available
__device__ __forceinline__ void copy16B(const __bf16* g, __bf16* l) {
#if HAVE_ASYNC_LDS
    __builtin_amdgcn_global_load_async_to_lds_b128((v4i*)g, (v4i*)l, 0, 0);
#else
    *(uint4*)l = *(const uint4*)g;
#endif
}

__device__ __forceinline__ void asyncFence() {
#if HAVE_ASYNC_LDS
    __builtin_amdgcn_s_wait_asynccnt(0);
#endif
}

// ---------------------------------------------------------------- init
__global__ void k_init(const float* __restrict__ hx, float* __restrict__ hbuf,
                       unsigned* __restrict__ bar) {
    int i = blockIdx.x * blockDim.x + threadIdx.x;
    if (i < HID) {
        hbuf[i]       = hx[i];   // ping buffer = h0
        hbuf[HID + i] = 0.0f;    // pong buffer
    }
    if (i == 0) { bar[0] = 0u; bar[1] = 0u; }
}

// ------------------------------------------------- embed lookup -> bf16
__global__ void k_embed(const int* __restrict__ inputs,
                        const float* __restrict__ emb,
                        __bf16* __restrict__ xb) {
    int idx = blockIdx.x * blockDim.x + threadIdx.x;   // SEQ*HID exact
    int t = idx >> 11;            // / HID
    int k = idx & (HID - 1);
    xb[idx] = (__bf16)emb[(size_t)inputs[t] * HID + k];
}

// ------------------------------------------------- fp32 -> bf16 weights
__global__ void k_cvt(const float* __restrict__ w, __bf16* __restrict__ wb, int n) {
    int i = blockIdx.x * blockDim.x + threadIdx.x;
    if (i < n) wb[i] = (__bf16)w[i];
}

// ---------------------------------------------------------------- GEMM
// gi[512,6144] = X_bf16[512,2048] @ W_ih^T + b_ih  via v_wmma_f32_16x16x32_bf16
// Block = (mTile, 512-column group): A strip staged in LDS (async copy) and
// shared by 8 waves; B fragments double-buffered in registers to overlap the
// L2-resident B stream with the WMMA pipe.
__global__ void __launch_bounds__(256) k_gemm_gi(const __bf16* __restrict__ xb,
                                                 const __bf16* __restrict__ wb,
                                                 const float*  __restrict__ bias,
                                                 float* __restrict__ gi) {
    constexpr int NT    = 4;      // N-tiles per wave
    constexpr int KTILE = 1024;   // staged K per phase (16*KTILE*2 = 32 KB LDS)
    __shared__ __align__(16) __bf16 sA[16 * KTILE];

    const int wave = threadIdx.x >> 5;
    const int lane = threadIdx.x & 31;
    const int mTile    = blockIdx.x / 12;   // 0..31
    const int colGroup = blockIdx.x % 12;   // 0..11 (512 cols each)
    const int n0 = colGroup * 512 + wave * 64;

    const int ml = lane & 15;     // matrix row/col owned by this lane
    const int kg = lane >> 4;     // k-group (ISA fragment layout)

    const __bf16* gA    = xb + (size_t)(mTile * 16) * HID;
    const __bf16* brow0 = wb + (size_t)(n0 + ml) * HID + kg * 8;
    const __bf16* aLds  = sA + ml * KTILE + kg * 8;

    v8f acc[NT] = {};
    FragU b0[NT], b1[NT];

    auto loadB = [&](FragU* b, int kglob) {
#pragma unroll
        for (int nt = 0; nt < NT; nt++) {
            const __bf16* br = brow0 + (size_t)nt * 16 * HID + kglob;
            b[nt].q[0] = *(const uint4*)(br);
            b[nt].q[1] = *(const uint4*)(br + 16);
        }
    };

    loadB(b0, 0);   // prime the pipeline

    for (int phase = 0; phase < HID / KTILE; phase++) {
        if (phase) __syncthreads();          // previous phase readers done
        // Fill sA with A[mTile*16 .. +16, phase*KTILE .. +KTILE): 2048 chunks
#pragma unroll
        for (int i = 0; i < 8; i++) {
            const int c    = threadIdx.x + i * 256;  // 16B chunk id
            const int row  = c >> 7;                 // 128 chunks per row
            const int col8 = c & 127;
            copy16B(gA + (size_t)row * HID + phase * KTILE + col8 * 8,
                    sA + row * KTILE + col8 * 8);
        }
        asyncFence();
        __syncthreads();

        for (int kb = 0; kb < KTILE; kb += 64) {
            const int kglob = phase * KTILE + kb;
            FragU a0, a1;
            a0.q[0] = *(const uint4*)(aLds + kb);
            a0.q[1] = *(const uint4*)(aLds + kb + 16);
            loadB(b1, kglob + 32);                       // in-flight during wmma(b0)
#pragma unroll
            for (int nt = 0; nt < NT; nt++)
                acc[nt] = __builtin_amdgcn_wmma_f32_16x16x32_bf16(
                    false, a0.v, false, b0[nt].v, (short)0, acc[nt], false, false);
            a1.q[0] = *(const uint4*)(aLds + kb + 32);
            a1.q[1] = *(const uint4*)(aLds + kb + 48);
            if (kglob + 64 < HID) loadB(b0, kglob + 64); // in-flight during wmma(b1)
#pragma unroll
            for (int nt = 0; nt < NT; nt++)
                acc[nt] = __builtin_amdgcn_wmma_f32_16x16x32_bf16(
                    false, a1.v, false, b1[nt].v, (short)0, acc[nt], false, false);
        }
    }

    // D layout: element v of lane l -> D[(l>>4)*8 + v, l&15]
    const int mBase = mTile * 16 + kg * 8;
#pragma unroll
    for (int nt = 0; nt < NT; nt++) {
        const int n = n0 + nt * 16 + ml;
        const float bv = bias[n];
#pragma unroll
        for (int v = 0; v < 8; v++)
            gi[(size_t)(mBase + v) * G3 + n] = acc[nt][v] + bv;
    }
}

// ---------------------------------------------------------- recurrence
__device__ __forceinline__ float waveReduce(float x) {
#pragma unroll
    for (int off = 16; off > 0; off >>= 1) x += __shfl_xor(x, off, 32);
    return x;
}

__device__ __forceinline__ void grid_barrier(unsigned* bar) {
    __threadfence();          // make this block's h writes visible device-wide
    __syncthreads();
    if (threadIdx.x == 0) {
        volatile unsigned* vgen = bar + 1;
        unsigned g = *vgen;
        __threadfence();
        if (atomicAdd(bar, 1u) == GRU_BLOCKS - 1) {
            *bar = 0u;        // all blocks have arrived (and read g) — safe to reset
            __threadfence();
            *vgen = g + 1u;
        } else {
            while (*vgen == g) { __builtin_amdgcn_s_sleep(2); }
        }
    }
    __syncthreads();
}

__global__ void k_gru(const float* __restrict__ gi,
                      const __bf16* __restrict__ whh,
                      const float*  __restrict__ bhh,
                      float* __restrict__ hbuf,
                      unsigned* __restrict__ bar) {
    const int lane   = threadIdx.x & 31;
    const int waveId = (blockIdx.x * blockDim.x + threadIdx.x) >> 5;  // 0..511

    for (int t = 0; t < SEQ; t++) {
        const float* hin  = hbuf + (t & 1) * HID;
        float*       hout = hbuf + ((t + 1) & 1) * HID;
        const float* git  = gi + (size_t)t * G3;
#pragma unroll
        for (int jj = 0; jj < JPW; jj++) {
            const int j = waveId * JPW + jj;
            const __bf16* wr = whh + (size_t)j * HID;
            const __bf16* wz = whh + (size_t)(j + HID) * HID;
            const __bf16* wn = whh + (size_t)(j + 2 * HID) * HID;
            float sr = 0.f, sz = 0.f, sn = 0.f;
            for (int k = lane * 4; k < HID; k += 128) {
                __builtin_prefetch(wr + k + 512, 0, 1);
                const float4 h4 = *(const float4*)(hin + k);
                const v4bf r4 = *(const v4bf*)(wr + k);
                const v4bf z4 = *(const v4bf*)(wz + k);
                const v4bf n4 = *(const v4bf*)(wn + k);
                sr = fmaf((float)r4[0], h4.x, sr); sr = fmaf((float)r4[1], h4.y, sr);
                sr = fmaf((float)r4[2], h4.z, sr); sr = fmaf((float)r4[3], h4.w, sr);
                sz = fmaf((float)z4[0], h4.x, sz); sz = fmaf((float)z4[1], h4.y, sz);
                sz = fmaf((float)z4[2], h4.z, sz); sz = fmaf((float)z4[3], h4.w, sz);
                sn = fmaf((float)n4[0], h4.x, sn); sn = fmaf((float)n4[1], h4.y, sn);
                sn = fmaf((float)n4[2], h4.z, sn); sn = fmaf((float)n4[3], h4.w, sn);
            }
            const float hr = waveReduce(sr) + bhh[j];
            const float hz = waveReduce(sz) + bhh[j + HID];
            const float hn = waveReduce(sn) + bhh[j + 2 * HID];
            if (lane == 0) {
                const float r = 1.0f / (1.0f + __expf(-(git[j] + hr)));
                const float z = 1.0f / (1.0f + __expf(-(git[j + HID] + hz)));
                const float n = tanhf(git[j + 2 * HID] + r * hn);
                hout[j] = (1.0f - z) * n + z * hin[j];
            }
        }
        grid_barrier(bar);
    }
}

// ------------------------------------------------------------- head
__global__ void k_out(const float* __restrict__ hbuf,
                      const float* __restrict__ w_out,
                      const float* __restrict__ b_out,
                      float* __restrict__ out) {
    __shared__ float s0[8], s1[8];
    const float* h = hbuf;     // SEQ even -> final state in ping buffer
    float a0 = 0.f, a1 = 0.f;
    for (int j = threadIdx.x; j < HID; j += 256) {
        const float hv = h[j];
        a0 = fmaf(hv, w_out[j], a0);
        a1 = fmaf(hv, w_out[HID + j], a1);
    }
    a0 = waveReduce(a0);
    a1 = waveReduce(a1);
    const int wave = threadIdx.x >> 5, lane = threadIdx.x & 31;
    if (lane == 0) { s0[wave] = a0; s1[wave] = a1; }
    __syncthreads();
    if (threadIdx.x == 0) {
        float l0 = b_out[0], l1 = b_out[1];
#pragma unroll
        for (int w = 0; w < 8; w++) { l0 += s0[w]; l1 += s1[w]; }
        const float m   = fmaxf(l0, l1);
        const float lse = m + logf(expf(l0 - m) + expf(l1 - m));
        out[0] = l0 - lse;
        out[1] = l1 - lse;
    }
}

// ---------------------------------------------------------------- host
static inline size_t align256(size_t x) { return (x + 255) & ~(size_t)255; }

extern "C" void kernel_launch(void* const* d_in, const int* in_sizes, int n_in,
                              void* d_out, int out_size, void* d_ws, size_t ws_size,
                              hipStream_t stream) {
    (void)in_sizes; (void)n_in; (void)out_size; (void)ws_size;
    const int*   inputs = (const int*)  d_in[0];
    const float* hx     = (const float*)d_in[1];
    const float* emb    = (const float*)d_in[2];
    const float* w_ih   = (const float*)d_in[3];
    const float* w_hh   = (const float*)d_in[4];
    const float* b_ih   = (const float*)d_in[5];
    const float* b_hh   = (const float*)d_in[6];
    const float* w_out  = (const float*)d_in[7];
    const float* b_out  = (const float*)d_in[8];
    float* out = (float*)d_out;

    char* ws = (char*)d_ws;
    size_t off = 0;
    __bf16*   xb   = (__bf16*)(ws + off); off = align256(off + (size_t)SEQ * HID * 2);
    __bf16*   wihb = (__bf16*)(ws + off); off = align256(off + (size_t)G3 * HID * 2);
    __bf16*   whhb = (__bf16*)(ws + off); off = align256(off + (size_t)G3 * HID * 2);
    float*    gi   = (float*) (ws + off); off = align256(off + (size_t)SEQ * G3 * 4);
    float*    hbuf = (float*) (ws + off); off = align256(off + (size_t)2 * HID * 4);
    unsigned* bar  = (unsigned*)(ws + off);

    k_init<<<(HID + 255) / 256, 256, 0, stream>>>(hx, hbuf, bar);
    k_embed<<<(SEQ * HID) / 256, 256, 0, stream>>>(inputs, emb, xb);
    k_cvt<<<(G3 * HID) / 256, 256, 0, stream>>>(w_ih, wihb, G3 * HID);
    k_cvt<<<(G3 * HID) / 256, 256, 0, stream>>>(w_hh, whhb, G3 * HID);
    // 32 M-tiles * 12 column-groups (512 cols, 8 waves * 4 N-tiles) = 384 blocks
    k_gemm_gi<<<384, GRU_TPB, 0, stream>>>(xb, wihb, b_ih, gi);
    k_gru<<<GRU_BLOCKS, GRU_TPB, 0, stream>>>(gi, whhb, b_hh, hbuf, bar);
    k_out<<<1, 256, 0, stream>>>(hbuf, w_out, b_out, out);
}
